// ODEEncoder_47493748359256
// MI455X (gfx1250) — matrix operand order repaired and checked by reading, not compiled, not run on hardware
//
#include <hip/hip_runtime.h>
#include <hip/hip_bf16.h>
#include <cmath>

// Problem constants (match reference)
#define Bsz 256
#define Tsz 512
#define DIN 64
#define Zs  32
#define Hs  64
#define TM1 511   // T-1

typedef __attribute__((ext_vector_type(16))) _Float16 v16h;
typedef __attribute__((ext_vector_type(8)))  _Float16 v8h;
typedef __attribute__((ext_vector_type(8)))  float    v8f;
typedef __attribute__((ext_vector_type(4)))  float    v4f;

// ---------- WMMA fragment helpers (layouts per CDNA5 ISA 7.12.2, wave32) ----

// A-matrix 16x32 f16: lane l holds row m = l&15; halves j=0..7 -> K=koff+j,
// j=8..15 -> K=koff+16+(j-8), with koff = (l>=16)?8:0.  => two contiguous
// 8-half (16B) runs per lane: perfect for B128 loads from row-major tiles.
__device__ __forceinline__ int kpatA(int lane, int j) {
  return (j < 8 ? j : j + 8) + ((lane >= 16) ? 8 : 0);
}

// B-matrix 32x16 f16 (K x N): lane n = lane&15; same K pattern as A (kpatA).
// W row-major (N_out x K_in): B[k][n] = W[n*ld+k].
__device__ __forceinline__ v16h load_bfrag(const float* __restrict__ W,
                                           int ld, int n, int k0, int lane) {
  v16h b;
  const int kh = (lane >= 16) ? 8 : 0;
#pragma unroll
  for (int j = 0; j < 16; ++j) b[j] = (_Float16)W[n * ld + k0 + kh + (j < 8 ? j : j + 8)];
  return b;
}

// Gather A fragment from a row-major f16 LDS tile: two ds_load_b128 per lane.
__device__ __forceinline__ v16h gatherA(const _Float16* tile, int ld, int m,
                                        int k0, int lane) {
  const int koff = (lane >= 16) ? 8 : 0;
  const _Float16* p = tile + m * ld + k0 + koff;
  const v8h lo = *(const v8h*)p;
  const v8h hi = *(const v8h*)(p + 16);
  v16h a;
#pragma unroll
  for (int j = 0; j < 8; ++j) { a[j] = lo[j]; a[8 + j] = hi[j]; }
  return a;
}

// Gather A fragment from a row-major f32 LDS tile (4x ds_load_b128 + cvt).
__device__ __forceinline__ v16h gatherA_f32(const float* tile, int ld, int m,
                                            int k0, int lane) {
  const int koff = (lane >= 16) ? 8 : 0;
  const float* p = tile + m * ld + k0 + koff;
  const v4f a0 = *(const v4f*)p;
  const v4f a1 = *(const v4f*)(p + 4);
  const v4f b0 = *(const v4f*)(p + 16);
  const v4f b1 = *(const v4f*)(p + 20);
  v16h a;
#pragma unroll
  for (int j = 0; j < 4; ++j) {
    a[j]      = (_Float16)a0[j];
    a[4 + j]  = (_Float16)a1[j];
    a[8 + j]  = (_Float16)b0[j];
    a[12 + j] = (_Float16)b1[j];
  }
  return a;
}

__device__ __forceinline__ v8f wmma_f16(v16h a, v16h b, v8f c) {
  return __builtin_amdgcn_wmma_f32_16x16x32_f16(
      /*neg_a=*/false, a, /*neg_b=*/false, b,
      /*c_mod=*/(short)0, c, /*reuse_a=*/false, /*reuse_b=*/false);
}

__device__ __forceinline__ v8f bsplat(float v) {
  v8f c;
#pragma unroll
  for (int r = 0; r < 8; ++r) c[r] = v;
  return c;
}

// Branchless fast tanh: 1 - 2/(e^{2x}+1); saturates to +/-1 at the extremes.
__device__ __forceinline__ float tanh_fast(float x) {
  const float e = __expf(2.0f * x);
  return 1.0f - 2.0f * __builtin_amdgcn_rcpf(e + 1.0f);
}

__device__ __forceinline__ float eluf(float v) {
  return v > 0.0f ? v : (__expf(v) - 1.0f);
}

// Wave-local LDS fence (private per-wave scratch; no block barrier needed).
__device__ __forceinline__ void wave_lds_sync() {
  __builtin_amdgcn_wave_barrier();
  asm volatile("s_wait_dscnt 0x0" ::: "memory");
  __builtin_amdgcn_wave_barrier();
}

// ============================================================================
// Kernel 1: reversed 2-layer tanh RNN. Grid 16 blocks x 64 threads.
// Block owns 16 batch rows; wave w owns N-tile w of Z=32.
// x tiles (16x64 f32, 4KB) are double-buffered in LDS via
// global_load_async_to_lds_b128 and pipelined one timestep ahead (ASYNCcnt).
// ============================================================================
__device__ __forceinline__ void issue_x_tile_async(const float* __restrict__ x,
                                                   float* ldsbuf, int b0,
                                                   int t, int tid) {
#pragma unroll
  for (int i = 0; i < 4; ++i) {
    const int idx = tid * 4 + i * 256;       // element in 16x64 tile
    const int row = idx >> 6, col = idx & 63;
    const unsigned goff =
        (unsigned)((((unsigned)(b0 + row) * Tsz + (unsigned)t) * DIN + col) *
                   sizeof(float));
    const unsigned laddr = (unsigned)(size_t)(ldsbuf + idx);  // low 32b = LDS off
    asm volatile("global_load_async_to_lds_b128 %0, %1, %2"
                 :: "v"(laddr), "v"(goff), "s"(x) : "memory");
  }
}

__global__ __launch_bounds__(64) void rnn_kernel(
    const float* __restrict__ x,
    const float* __restrict__ Wi0, const float* __restrict__ Wh0,
    const float* __restrict__ bi0, const float* __restrict__ bh0,
    const float* __restrict__ Wi1, const float* __restrict__ Wh1,
    const float* __restrict__ bi1, const float* __restrict__ bh1,
    const float* __restrict__ h0,
    float* __restrict__ rnn_ws, float* __restrict__ out)
{
  __shared__ __align__(16) _Float16 hbuf[16 * Zs];
  __shared__ __align__(16) float    xbuf[2][16 * DIN];   // double buffer, 8KB
  const int tid  = threadIdx.x;
  const int lane = tid & 31;
  const int wave = tid >> 5;                // 0..1 -> N tile
  const int b0   = blockIdx.x * 16;
  const int nlo  = lane & 15;
  const int ncol = wave * 16 + nlo;         // output column in Z
  const int m    = lane & 15;               // A-matrix row within tile
  const int rhi  = (lane >= 16) ? 8 : 0;    // C-layout row offset

  // Resident weight fragments (f16).
  const v16h bWi0a = load_bfrag(Wi0, DIN, ncol, 0,  lane);
  const v16h bWi0b = load_bfrag(Wi0, DIN, ncol, 32, lane);
  const v16h bWh0  = load_bfrag(Wh0, Zs,  ncol, 0,  lane);
  const v16h bWi1  = load_bfrag(Wi1, Zs,  ncol, 0,  lane);
  const v16h bWh1  = load_bfrag(Wh1, Zs,  ncol, 0,  lane);
  const float bias0 = bi0[ncol] + bh0[ncol];
  const float bias1 = bi1[ncol] + bh1[ncol];

  // h state kept as A fragments (h0 is (L,1,Z), broadcast across rows).
  v16h a_h0, a_h1;
#pragma unroll
  for (int j = 0; j < 16; ++j) {
    const int k = kpatA(lane, j);
    a_h0[j] = (_Float16)h0[k];
    a_h1[j] = (_Float16)h0[Zs + k];
  }

  // Prologue: stage x tile for s=0 (t = T-1).
  issue_x_tile_async(x, xbuf[0], b0, Tsz - 1, tid);

  for (int s = 0; s < Tsz; ++s) {
    const int t = Tsz - 1 - s;                 // reversed time
    const float* xtile = xbuf[s & 1];

    // Pipeline: kick off next step's tile, then wait for the current one
    // (<=4 outstanding == next tile only). Completion is in issue order.
    if (s + 1 < Tsz) {
      issue_x_tile_async(x, xbuf[(s + 1) & 1], b0, t - 1, tid);
      asm volatile("s_wait_asynccnt 0x4" ::: "memory");
    } else {
      asm volatile("s_wait_asynccnt 0x0" ::: "memory");
    }
    __syncthreads();   // x tile visible to both waves; prev hbuf gathers done

    const v16h ax0 = gatherA_f32(xtile, DIN, m, 0,  lane);
    const v16h ax1 = gatherA_f32(xtile, DIN, m, 32, lane);

    // Layer 0: pre = x@Wi0^T + h0@Wh0^T + (bi0+bh0)
    v8f acc0 = bsplat(bias0);
    acc0 = wmma_f16(ax0,  bWi0a, acc0);
    acc0 = wmma_f16(ax1,  bWi0b, acc0);
    acc0 = wmma_f16(a_h0, bWh0,  acc0);
#pragma unroll
    for (int r = 0; r < 8; ++r) acc0[r] = tanh_fast(acc0[r]);

#pragma unroll
    for (int r = 0; r < 8; ++r)
      hbuf[(r + rhi) * Zs + ncol] = (_Float16)acc0[r];
    __syncthreads();
    a_h0 = gatherA(hbuf, Zs, m, 0, lane);      // full-Z A fragment of new h0

    // Layer 1
    v8f acc1 = bsplat(bias1);
    acc1 = wmma_f16(a_h0, bWi1, acc1);
    acc1 = wmma_f16(a_h1, bWh1, acc1);
#pragma unroll
    for (int r = 0; r < 8; ++r) acc1[r] = tanh_fast(acc1[r]);

    __syncthreads();   // a_h0 gathers done before hbuf overwrite
#pragma unroll
    for (int r = 0; r < 8; ++r)
      hbuf[(r + rhi) * Zs + ncol] = (_Float16)acc1[r];
    __syncthreads();
    a_h1 = gatherA(hbuf, Zs, m, 0, lane);

    // rnn_out[:, t, :] to workspace; t == T-1 also goes straight to output.
#pragma unroll
    for (int r = 0; r < 8; ++r) {
      const size_t row = (size_t)(b0 + r + rhi);
      const float v = acc1[r];
      rnn_ws[(row * Tsz + t) * Zs + ncol] = v;
      if (t == Tsz - 1) out[(row * Tsz + t) * Zs + ncol] = v;
    }
  }
}

// ============================================================================
// Kernel 2: RK4 over f = fc3(elu(fc2(elu(fc1(y))))), 9 steps.
// Each wave owns a 16-row tile and a private LDS scratch (wave-local fences).
// ============================================================================
struct OdeW {
  v16h w1[4];       // fc1: (64,32)  -> 4 N-tiles, K=32
  v16h w2[4][2];    // fc2: (64,64)  -> 4 N-tiles x 2 K-chunks
  v16h w3[2][2];    // fc3: (32,64)  -> 2 N-tiles x 2 K-chunks
  float b1[4], b2[4], b3[2];
};

__device__ __forceinline__ void mlp_f(const v8f yin[2], v8f fout[2],
                                      _Float16* sm, const OdeW& W,
                                      int lane, int m, int nlo, int rhi)
{
  // ---- stage y (16x32 f32 C-layout) -> LDS f16, rebuild A fragment
  wave_lds_sync();   // prior reads of sm complete before overwrite
#pragma unroll
  for (int tc = 0; tc < 2; ++tc)
#pragma unroll
    for (int r = 0; r < 8; ++r)
      sm[(r + rhi) * Zs + tc * 16 + nlo] = (_Float16)yin[tc][r];
  wave_lds_sync();
  const v16h a_y = gatherA(sm, Zs, m, 0, lane);

  // ---- fc1 + ELU  (16x32)x(32x64)
  v8f h1[4];
#pragma unroll
  for (int nt = 0; nt < 4; ++nt) {
    h1[nt] = wmma_f16(a_y, W.w1[nt], bsplat(W.b1[nt]));
#pragma unroll
    for (int r = 0; r < 8; ++r) h1[nt][r] = eluf(h1[nt][r]);
  }

  wave_lds_sync();
#pragma unroll
  for (int nt = 0; nt < 4; ++nt)
#pragma unroll
    for (int r = 0; r < 8; ++r)
      sm[(r + rhi) * Hs + nt * 16 + nlo] = (_Float16)h1[nt][r];
  wave_lds_sync();
  const v16h a1k0 = gatherA(sm, Hs, m, 0,  lane);
  const v16h a1k1 = gatherA(sm, Hs, m, 32, lane);

  // ---- fc2 + ELU  (16x64)x(64x64)
  v8f h2[4];
#pragma unroll
  for (int nt = 0; nt < 4; ++nt) {
    v8f c = bsplat(W.b2[nt]);
    c = wmma_f16(a1k0, W.w2[nt][0], c);
    c = wmma_f16(a1k1, W.w2[nt][1], c);
#pragma unroll
    for (int r = 0; r < 8; ++r) c[r] = eluf(c[r]);
    h2[nt] = c;
  }

  wave_lds_sync();
#pragma unroll
  for (int nt = 0; nt < 4; ++nt)
#pragma unroll
    for (int r = 0; r < 8; ++r)
      sm[(r + rhi) * Hs + nt * 16 + nlo] = (_Float16)h2[nt][r];
  wave_lds_sync();
  const v16h a2k0 = gatherA(sm, Hs, m, 0,  lane);
  const v16h a2k1 = gatherA(sm, Hs, m, 32, lane);

  // ---- fc3  (16x64)x(64x32)
#pragma unroll
  for (int nt = 0; nt < 2; ++nt) {
    v8f c = bsplat(W.b3[nt]);
    c = wmma_f16(a2k0, W.w3[nt][0], c);
    c = wmma_f16(a2k1, W.w3[nt][1], c);
    fout[nt] = c;
  }
}

__global__ __launch_bounds__(256) void ode_kernel(
    const float* __restrict__ rnn_ws,
    const float* __restrict__ fc1w, const float* __restrict__ fc1b,
    const float* __restrict__ fc2w, const float* __restrict__ fc2b,
    const float* __restrict__ fc3w, const float* __restrict__ fc3b,
    float* __restrict__ out)
{
  __shared__ __align__(16) _Float16 abuf[8][16 * Hs];   // 2KB per wave
  const int lane = threadIdx.x & 31;
  const int wave = threadIdx.x >> 5;
  const int nlo  = lane & 15;
  const int m    = lane & 15;
  const int rhi  = (lane >= 16) ? 8 : 0;
  _Float16* sm = abuf[wave];

  OdeW W;
#pragma unroll
  for (int nt = 0; nt < 4; ++nt) {
    W.w1[nt]    = load_bfrag(fc1w, Zs, nt * 16 + nlo, 0,  lane);
    W.w2[nt][0] = load_bfrag(fc2w, Hs, nt * 16 + nlo, 0,  lane);
    W.w2[nt][1] = load_bfrag(fc2w, Hs, nt * 16 + nlo, 32, lane);
    W.b1[nt] = fc1b[nt * 16 + nlo];
    W.b2[nt] = fc2b[nt * 16 + nlo];
  }
#pragma unroll
  for (int nt = 0; nt < 2; ++nt) {
    W.w3[nt][0] = load_bfrag(fc3w, Hs, nt * 16 + nlo, 0,  lane);
    W.w3[nt][1] = load_bfrag(fc3w, Hs, nt * 16 + nlo, 32, lane);
    W.b3[nt] = fc3b[nt * 16 + nlo];
  }

  // This wave's 16 rows of the flattened (B*(T-1), 32) state.
  const size_t row0 = ((size_t)blockIdx.x * 8 + wave) * 16;

  // y0 = rnn_out[:, 1:, :] in C layout (2 column tiles of v8f).
  v8f y[2];
#pragma unroll
  for (int tc = 0; tc < 2; ++tc)
#pragma unroll
    for (int r = 0; r < 8; ++r) {
      const size_t rr = row0 + (size_t)(r + rhi);
      const size_t b = rr / TM1, j = rr % TM1;
      y[tc][r] = rnn_ws[(b * Tsz + j + 1) * Zs + tc * 16 + nlo];
    }

  const float hstep = 0.1f / 10.0f;     // DT / DISC
  v8f k[2], sum[2], arg[2];
  for (int it = 0; it < 9; ++it) {      // DISC - 1 RK4 steps
    mlp_f(y, k, sm, W, lane, m, nlo, rhi);              // k1
#pragma unroll
    for (int tc = 0; tc < 2; ++tc)
#pragma unroll
      for (int r = 0; r < 8; ++r) {
        sum[tc][r] = k[tc][r];
        arg[tc][r] = y[tc][r] + 0.5f * hstep * k[tc][r];
      }
    mlp_f(arg, k, sm, W, lane, m, nlo, rhi);            // k2
#pragma unroll
    for (int tc = 0; tc < 2; ++tc)
#pragma unroll
      for (int r = 0; r < 8; ++r) {
        sum[tc][r] += 2.0f * k[tc][r];
        arg[tc][r] = y[tc][r] + 0.5f * hstep * k[tc][r];
      }
    mlp_f(arg, k, sm, W, lane, m, nlo, rhi);            // k3
#pragma unroll
    for (int tc = 0; tc < 2; ++tc)
#pragma unroll
      for (int r = 0; r < 8; ++r) {
        sum[tc][r] += 2.0f * k[tc][r];
        arg[tc][r] = y[tc][r] + hstep * k[tc][r];
      }
    mlp_f(arg, k, sm, W, lane, m, nlo, rhi);            // k4
#pragma unroll
    for (int tc = 0; tc < 2; ++tc)
#pragma unroll
      for (int r = 0; r < 8; ++r)
        y[tc][r] += (hstep / 6.0f) * (sum[tc][r] + k[tc][r]);
  }

  // out[:, 0:T-1, :] = yT
#pragma unroll
  for (int tc = 0; tc < 2; ++tc)
#pragma unroll
    for (int r = 0; r < 8; ++r) {
      const size_t rr = row0 + (size_t)(r + rhi);
      const size_t b = rr / TM1, j = rr % TM1;
      out[(b * Tsz + j) * Zs + tc * 16 + nlo] = y[tc][r];
    }
}

// ============================================================================
extern "C" void kernel_launch(void* const* d_in, const int* in_sizes, int n_in,
                              void* d_out, int out_size, void* d_ws, size_t ws_size,
                              hipStream_t stream) {
  const float* x   = (const float*)d_in[0];
  const float* Wi0 = (const float*)d_in[1];
  const float* Wh0 = (const float*)d_in[2];
  const float* bi0 = (const float*)d_in[3];
  const float* bh0 = (const float*)d_in[4];
  const float* Wi1 = (const float*)d_in[5];
  const float* Wh1 = (const float*)d_in[6];
  const float* bi1 = (const float*)d_in[7];
  const float* bh1 = (const float*)d_in[8];
  const float* h0  = (const float*)d_in[9];
  const float* f1w = (const float*)d_in[10];
  const float* f1b = (const float*)d_in[11];
  const float* f2w = (const float*)d_in[12];
  const float* f2b = (const float*)d_in[13];
  const float* f3w = (const float*)d_in[14];
  const float* f3b = (const float*)d_in[15];
  float* out = (float*)d_out;
  float* ws  = (float*)d_ws;   // B*T*Z floats = 16 MB of rnn_out scratch

  // Phase 1: reversed RNN -> rnn_out in workspace (+ last timestep to out).
  rnn_kernel<<<Bsz / 16, 64, 0, stream>>>(x, Wi0, Wh0, bi0, bh0,
                                          Wi1, Wh1, bi1, bh1, h0, ws, out);

  // Phase 2: RK4 MLP integrator over B*(T-1) rows.
  const int tiles  = (Bsz * TM1) / 16;   // 8176 (exact)
  const int blocks = tiles / 8;          // 1022 (exact)
  ode_kernel<<<blocks, 256, 0, stream>>>(ws, f1w, f1b, f2w, f2b, f3w, f3b, out);
}
